// FuzzyNeuralNework_49022756717024
// MI455X (gfx1250) — matrix operand order repaired and checked by reading, not compiled, not run on hardware
//
#include <hip/hip_runtime.h>
#include <hip/hip_bf16.h>

// ---------------------------------------------------------------------------
// FuzzyNeuralNetwork forward for MI455X (gfx1250, wave32, WMMA).
// B=8192, D=128, R=64, C=64.
//
// out[b,c] = sum_{r,d} (frs[b,r]*xn[b,d]) * w[r,d,c]  +  (frs @ bias)[b,c]
//
// GEMM kernel (v2): 64 blocks x 4 waves. Per rule, the 16KB packed-B block is
// streamed into LDS with global_load_async_to_lds_b128 (double-buffered,
// ASYNCcnt + workgroup barrier), shared by all 4 waves; each wave owns TWO
// 16-row batch tiles so every LDS B-fragment feeds two
// v_wmma_f32_16x16x32_bf16 ops (8x total B reuse vs v1 -> 64MB L2 traffic).
// Per-rule row rescale uses v_pk_mul_bf16 (one frs scalar per A lane).
// ---------------------------------------------------------------------------

typedef __attribute__((ext_vector_type(16))) __bf16 v16bf;
typedef __attribute__((ext_vector_type(8)))  float  v8f;

#define B_SZ 8192
#define D_SZ 128
#define R_SZ 64
#define C_SZ 64

union Frag {
    v16bf v;
    unsigned u[8];
    uint4 q[2];
};

__device__ inline Frag load_frag(const __bf16* p0, const __bf16* p1) {
    Frag f;
    f.q[0] = *(const uint4*)p0;   // 8 bf16: K-run 0
    f.q[1] = *(const uint4*)p1;   // 8 bf16: K-run 1
    return f;
}

// ---------------------------------------------------------------------------
// Kernel 1: BatchNorm batch statistics -> per-feature affine (a[d], bb[d])
// ---------------------------------------------------------------------------
__global__ __launch_bounds__(256) void fnn_bnstats(
    const float* __restrict__ x, const float* __restrict__ gamma,
    const float* __restrict__ beta, float* __restrict__ abuf,
    float* __restrict__ bbuf)
{
    __shared__ float s1[256];
    __shared__ float s2[256];
    const int d = blockIdx.x;
    const int tid = threadIdx.x;
    float sum = 0.f, sq = 0.f;
    for (int b = tid; b < B_SZ; b += 256) {
        float v = x[(size_t)b * D_SZ + d];
        sum += v;
        sq  += v * v;
    }
    s1[tid] = sum; s2[tid] = sq;
    __syncthreads();
    for (int st = 128; st > 0; st >>= 1) {
        if (tid < st) { s1[tid] += s1[tid + st]; s2[tid] += s2[tid + st]; }
        __syncthreads();
    }
    if (tid == 0) {
        float mean = s1[0] * (1.f / B_SZ);
        float var  = s2[0] * (1.f / B_SZ) - mean * mean;   // biased variance
        float a = rsqrtf(var + 1e-5f) * gamma[d];
        abuf[d] = a;
        bbuf[d] = beta[d] - mean * a;
    }
}

// ---------------------------------------------------------------------------
// Kernel 2: pre-pack weights/bias (f32) into bf16 WMMA B-fragment layout.
// B operand (32x16 KxN): lanes 0-15 hold K={0..7,16..23} at N=lane;
// lanes 16-31 hold K={8..15,24..31} at N=lane-16. Per-lane 16 bf16 stored
// contiguously: pw[(((r*4+k)*4+n)*32+lane)*16 + i].
// ---------------------------------------------------------------------------
__global__ __launch_bounds__(256) void fnn_pack(
    const float* __restrict__ w, const float* __restrict__ bias,
    __bf16* __restrict__ pw, __bf16* __restrict__ pb)
{
    const int NW = R_SZ * 4 * 4 * 32 * 16;   // 524288
    int idx = blockIdx.x * 256 + threadIdx.x;
    if (idx < NW) {
        int i    =  idx        & 15;
        int lane = (idx >> 4)  & 31;
        int n    = (idx >> 9)  & 3;
        int k    = (idx >> 11) & 3;
        int r    =  idx >> 13;
        int d    = k * 32 + ((lane >= 16) ? 8 : 0) + (i & 7) + ((i >= 8) ? 16 : 0);
        int c    = n * 16 + (lane & 15);
        pw[idx] = (__bf16)w[((size_t)r * D_SZ + d) * C_SZ + c];
    } else if (idx < NW + 2 * 4 * 32 * 16) {
        int j    = idx - NW;
        int i    =  j        & 15;
        int lane = (j >> 4)  & 31;
        int n    = (j >> 9)  & 3;
        int k2   =  j >> 11;                       // 0..1 (K = R = 64)
        int rr   = k2 * 32 + ((lane >= 16) ? 8 : 0) + (i & 7) + ((i >= 8) ? 16 : 0);
        int c    = n * 16 + (lane & 15);
        pb[j] = (__bf16)bias[rr * C_SZ + c];
    }
}

// ---------------------------------------------------------------------------
// Kernel 3: fuzzy layer (frs bf16 + xn bf16), one 64-thread block per row.
// ---------------------------------------------------------------------------
__global__ __launch_bounds__(64) void fnn_fuzzy(
    const float* __restrict__ x, const float* __restrict__ centers,
    const float* __restrict__ sigmas, const float* __restrict__ rmask,
    const float* __restrict__ abuf, const float* __restrict__ bbuf,
    __bf16* __restrict__ xnb, __bf16* __restrict__ frsb)
{
    __shared__ float xs[D_SZ];
    __shared__ float red[64];
    const int tid = threadIdx.x;
    const size_t b = blockIdx.x;

    for (int t = tid; t < D_SZ; t += 64) {
        float v = x[b * D_SZ + t];
        xs[t] = v;
        xnb[b * D_SZ + t] = (__bf16)(v * abuf[t] + bbuf[t]);
    }
    __syncthreads();

    const int r = tid;
    float logit = 0.f;
#pragma unroll 8
    for (int d = 0; d < D_SZ; ++d) {
        float c = centers[d * R_SZ + r];      // coalesced across threads
        float s = sigmas [d * R_SZ + r];
        float diff = xs[d] - c;
        logit -= (diff * diff) / (2.f * s * s);
    }
    float e = expf(logit) * rmask[r];
    red[tid] = e;
    __syncthreads();
    for (int st = 32; st > 0; st >>= 1) {
        if (tid < st) red[tid] += red[tid + st];
        __syncthreads();
    }
    frsb[b * R_SZ + r] = (__bf16)(e / (red[0] + 1e-10f));
}

// ---------------------------------------------------------------------------
// Kernel 4: WMMA GEMM with async-LDS B streaming.
// Block = 128 threads (4 waves); wave w owns batch rows [blk*128+w*32, +32)
// as two 16-row tiles. A-fragment layout (16-bit A 16x32): lane<16 holds row
// M=lane with K-runs {k*32+0..7, k*32+16..23}; lane>=16 holds runs {+8,+24}.
// ---------------------------------------------------------------------------
__global__ __launch_bounds__(128) void fnn_gemm(
    const __bf16* __restrict__ xnb, const __bf16* __restrict__ frsb,
    const __bf16* __restrict__ pw, const __bf16* __restrict__ pb,
    float* __restrict__ out)
{
    __shared__ uint4 sB[2][1024];           // double-buffered 2 x 16KB
    const int tid  = threadIdx.x;
    const int lane = tid & 31;
    const int wave = tid >> 5;
    const int m    = lane & 15;
    const int off  = (lane < 16) ? 0 : 8;   // K-run offset within 32-chunk
    const int hi   = (lane < 16) ? 0 : 8;   // C/D row offset

    const int rows0 = blockIdx.x * 128 + wave * 32;   // tile 0
    const int rows1 = rows0 + 16;                     // tile 1

    // A fragments for both tiles (rule-independent).
    Frag A0[4], A1[4];
    const __bf16* xr0 = xnb + (size_t)(rows0 + m) * D_SZ;
    const __bf16* xr1 = xnb + (size_t)(rows1 + m) * D_SZ;
#pragma unroll
    for (int k = 0; k < 4; ++k) {
        A0[k] = load_frag(xr0 + k * 32 + off, xr0 + k * 32 + 16 + off);
        A1[k] = load_frag(xr1 + k * 32 + off, xr1 + k * 32 + 16 + off);
    }

    const __bf16* fr0 = frsb + (size_t)(rows0 + m) * R_SZ;
    const __bf16* fr1 = frsb + (size_t)(rows1 + m) * R_SZ;

    // Accumulator init = frs @ bias (K = 64 -> 2 k-steps), via global loads.
    v8f acc0[4], acc1[4];
#pragma unroll
    for (int n = 0; n < 4; ++n)
#pragma unroll
        for (int j = 0; j < 8; ++j) { acc0[n][j] = 0.f; acc1[n][j] = 0.f; }
    {
        Frag FA0[2], FA1[2];
#pragma unroll
        for (int k2 = 0; k2 < 2; ++k2) {
            FA0[k2] = load_frag(fr0 + k2 * 32 + off, fr0 + k2 * 32 + 16 + off);
            FA1[k2] = load_frag(fr1 + k2 * 32 + off, fr1 + k2 * 32 + 16 + off);
        }
#pragma unroll
        for (int k2 = 0; k2 < 2; ++k2)
#pragma unroll
            for (int n = 0; n < 4; ++n) {
                const __bf16* bp = pb + (size_t)(((k2 * 4 + n) * 32) + lane) * 16;
                Frag Bf = load_frag(bp, bp + 8);
                acc0[n] = __builtin_amdgcn_wmma_f32_16x16x32_bf16(
                    false, FA0[k2].v, false, Bf.v, (short)0, acc0[n], false, false);
                acc1[n] = __builtin_amdgcn_wmma_f32_16x16x32_bf16(
                    false, FA1[k2].v, false, Bf.v, (short)0, acc1[n], false, false);
            }
    }

    // Async streamer: rule r's 16KB packed-B block -> sB[p].
    // 128 threads x 8 x b128 (consecutive threads -> consecutive 16B).
    auto issue_rule = [&](int r, int p) {
        unsigned lbase = (unsigned)(uintptr_t)(&sB[p][0]);
        const __bf16* src = pw + (size_t)r * (4 * 4 * 32 * 16);
#pragma unroll
        for (int j = 0; j < 8; ++j) {
            unsigned loff = lbase + (unsigned)((j * 128 + tid) * 16);
            unsigned long long g =
                (unsigned long long)(uintptr_t)(src + (size_t)(j * 128 + tid) * 8);
            asm volatile("global_load_async_to_lds_b128 %0, %1, off"
                         :: "v"(loff), "v"(g) : "memory");
        }
    };

    issue_rule(0, 0);

    for (int r = 0; r < R_SZ; ++r) {
        const int p = r & 1;
        asm volatile("s_wait_asynccnt 0x0" ::: "memory");
        __syncthreads();   // buf p filled by all waves; buf p^1 fully consumed
        if (r + 1 < R_SZ) issue_rule(r + 1, p ^ 1);

        unsigned short f0 = __builtin_bit_cast(unsigned short, fr0[r]);
        unsigned short f1 = __builtin_bit_cast(unsigned short, fr1[r]);
        unsigned fr20 = ((unsigned)f0 << 16) | (unsigned)f0;
        unsigned fr21 = ((unsigned)f1 << 16) | (unsigned)f1;

#pragma unroll
        for (int k = 0; k < 4; ++k) {
            Frag As0, As1;                     // per-rule row-rescaled A
#pragma unroll
            for (int i = 0; i < 8; ++i) {
                asm("v_pk_mul_bf16 %0, %1, %2"
                    : "=v"(As0.u[i]) : "v"(A0[k].u[i]), "v"(fr20));
                asm("v_pk_mul_bf16 %0, %1, %2"
                    : "=v"(As1.u[i]) : "v"(A1[k].u[i]), "v"(fr21));
            }
#pragma unroll
            for (int n = 0; n < 4; ++n) {      // B frag from LDS, used twice
                const uint4* bp = &sB[p][(size_t)(((k * 4 + n) * 32) + lane) * 2];
                Frag Bf;
                Bf.q[0] = bp[0];
                Bf.q[1] = bp[1];
                acc0[n] = __builtin_amdgcn_wmma_f32_16x16x32_bf16(
                    false, As0.v, false, Bf.v, (short)0, acc0[n], false, false);
                acc1[n] = __builtin_amdgcn_wmma_f32_16x16x32_bf16(
                    false, As1.v, false, Bf.v, (short)0, acc1[n], false, false);
            }
        }
    }

    // C/D layout: VGPR j, lanes 0-15 -> (M=j, N=lane); lanes 16-31 -> M=j+8.
#pragma unroll
    for (int n = 0; n < 4; ++n)
#pragma unroll
        for (int j = 0; j < 8; ++j) {
            out[(size_t)(rows0 + j + hi) * C_SZ + n * 16 + m] = acc0[n][j];
            out[(size_t)(rows1 + j + hi) * C_SZ + n * 16 + m] = acc1[n][j];
        }
}

// ---------------------------------------------------------------------------
// Workspace layout (256B-aligned offsets; ~4.01 MB total):
//   [0,512)    a[d] f32   | [512,1024)  bb[d] f32
//   [1024,+1MB)   frs bf16 [B,R]
//   [+1MB,+3MB)   xn  bf16 [B,D]
//   [+3MB,+4MB)   packed W bf16 | [+4MB,+8KB) packed bias bf16
// ---------------------------------------------------------------------------
extern "C" void kernel_launch(void* const* d_in, const int* in_sizes, int n_in,
                              void* d_out, int out_size, void* d_ws, size_t ws_size,
                              hipStream_t stream) {
    const float* x       = (const float*)d_in[0];
    const float* centers = (const float*)d_in[1];
    const float* sigmas  = (const float*)d_in[2];
    const float* weights = (const float*)d_in[3];
    const float* biases  = (const float*)d_in[4];
    const float* gamma   = (const float*)d_in[5];
    const float* beta    = (const float*)d_in[6];
    const float* rmask   = (const float*)d_in[7];

    char* ws = (char*)d_ws;
    float*  abuf = (float*)(ws + 0);
    float*  bbuf = (float*)(ws + 512);
    __bf16* frsb = (__bf16*)(ws + 1024);
    __bf16* xnb  = (__bf16*)(ws + 1024 + 1048576);
    __bf16* pw   = (__bf16*)(ws + 1024 + 1048576 + 2097152);
    __bf16* pb   = (__bf16*)(ws + 1024 + 1048576 + 2097152 + 1048576);
    float*  out  = (float*)d_out;

    fnn_bnstats<<<D_SZ, 256, 0, stream>>>(x, gamma, beta, abuf, bbuf);
    fnn_pack<<<(524288 + 4096 + 255) / 256, 256, 0, stream>>>(weights, biases, pw, pb);
    fnn_fuzzy<<<B_SZ, 64, 0, stream>>>(x, centers, sigmas, rmask, abuf, bbuf, xnb, frsb);
    fnn_gemm<<<B_SZ / 128, 128, 0, stream>>>(xnb, frsb, pw, pb, out);
}
